// Nnue_24352464569941
// MI455X (gfx1250) — compile-verified, matrix operand order checked by reading, not run on hardware
//
#include <hip/hip_runtime.h>
#include <hip/hip_bf16.h>
#include <math.h>

// ---- CDNA5 vector types ----
typedef __attribute__((ext_vector_type(16))) _Float16 v16h;
typedef __attribute__((ext_vector_type(8)))  _Float16 h8;
typedef __attribute__((ext_vector_type(4)))  _Float16 h4;
typedef __attribute__((ext_vector_type(8)))  float    v8f;
typedef __attribute__((ext_vector_type(4)))  float    f4;
typedef __attribute__((ext_vector_type(4)))  unsigned int u32x4;
typedef __attribute__((ext_vector_type(8)))  int      i32x8;
typedef __attribute__((ext_vector_type(4)))  int      i32x4;

#define F_DIM   36864          // NNUE features (K)
#define H1      256            // hidden width per perspective (N)
#define BATCH   4096
#define M_TOT   (2 * BATCH)    // both perspectives stacked -> 8192 rows
#define KSTEP   32             // K per v_wmma_f32_16x16x32_f16
#define NSTEPS  (F_DIM / KSTEP)
#define LDS_STRIDE 40          // halves; 80B row stride (TDM pad) -> conflict-free ds_load_b128

// TDM builtin arity differs: ROCm 7.2 (clang-22) = 5 args, clang-23 = 6 args.
#if defined(__clang_major__) && (__clang_major__ >= 23)
#define TDM_LOAD_TO_LDS(g0, g1, g2, g3) \
    __builtin_amdgcn_tensor_load_to_lds((g0), (g1), (g2), (g3), (i32x8)(0), 0)
#else
#define TDM_LOAD_TO_LDS(g0, g1, g2, g3) \
    __builtin_amdgcn_tensor_load_to_lds((g0), (g1), (g2), (g3), 0)
#endif

// ------------------------------------------------------------------
// Kernel 0: one-time W1 fp32 -> fp16 (18.9 MB, lives in 192MB L2 after)
// ------------------------------------------------------------------
__global__ void cvt_w1_kernel(const float* __restrict__ w1,
                              _Float16* __restrict__ w1h, int n4) {
    int i = blockIdx.x * blockDim.x + threadIdx.x;
    if (i >= n4) return;
    f4 v = ((const f4*)w1)[i];
    ((h4*)w1h)[i] = __builtin_convertvector(v, h4);
}

// ------------------------------------------------------------------
// Kernel 1: feature transformer  a[m, 0..255] = relu(x[m,:] @ W1^T + b1)
// M = 8192, N = 256, K = 36864.  WMMA f16 with f32 accumulate.
// Block: 256 threads (8 waves). Block tile: 128 M x 256 N.
// Wave tile: 16 M x 256 N (16 accumulators of 16x16 f32).
// W1^T k-slab staged in LDS by the Tensor Data Mover (double-buffered,
// TDM row padding 64B+16B -> 80B stride), x streamed from global.
// ------------------------------------------------------------------
__global__ __launch_bounds__(256, 1)
void nnue_layer1_kernel(const float* __restrict__ x,
                        const _Float16* __restrict__ w1h,
                        const float* __restrict__ b1,
                        _Float16* __restrict__ aout) {
    __shared__ _Float16 ldsW[2][H1 * LDS_STRIDE];  // 2 x 20 KB TDM destination
    __shared__ float    sb1[H1];

    const int tid  = threadIdx.x;
    const int lane = tid & 31;
    const int wave = tid >> 5;
    const int hlf  = lane >> 4;     // 0: lanes 0-15, 1: lanes 16-31
    const int l16  = lane & 15;
    const int kOff = hlf * 8;       // per ISA 16-bit A/B fragment layout

    if (tid < H1) sb1[tid] = b1[tid];

    const int mBase = blockIdx.x * 128 + wave * 16;
    const float* xrow = x + (size_t)(mBase + l16) * F_DIM;

    // 16 N-tiles of 16x16 f32 accumulators
    v8f acc[16];
    const v8f vzero = {0.f, 0.f, 0.f, 0.f, 0.f, 0.f, 0.f, 0.f};
#pragma unroll
    for (int nt = 0; nt < 16; ++nt) acc[nt] = vzero;

    // ---- Tensor DMA descriptor (D#) for one W1^T k-slab: 2D tile ----
    // tensor: 256 rows x 36864 fp16, row stride 36864; tile: 32 (K) x 256 (N)
    // pad: every 16 DWORDs (64B row) insert 4 DWORDs (16B) -> 80B LDS row stride
    const i32x8 g1 = {
        (int)((1u << 16) | (1u << 20) | (3u << 22) | (3u << 25)), // data_size=2B, pad_enable, pad_interval=16dw, pad_amount=4dw
        (int)((unsigned)(F_DIM & 0xFFFF) << 16),                  // tensor_dim0[15:0] (hi half of dword)
        (int)((F_DIM >> 16) | ((unsigned)H1 << 16)),              // tensor_dim0[31:16] | tensor_dim1[15:0]
        (int)((unsigned)KSTEP << 16),                             // tile_dim0 = 32
        (int)H1,                                                  // tile_dim1 = 256
        (int)F_DIM,                                               // tensor_dim0_stride lo
        0, 0
    };
    const i32x4 gz = {0, 0, 0, 0};                                // groups 2/3: unused (2D)
    const unsigned ldsOff0 = (unsigned)(uintptr_t)(&ldsW[0][0]);  // LDS byte offset (addr[31:0])
    const unsigned ldsOff1 = (unsigned)(uintptr_t)(&ldsW[1][0]);
    const unsigned long long gbase = (unsigned long long)(uintptr_t)w1h;

    auto issue_tdm = [&](unsigned ldsOff, int kb) {
        unsigned long long ga = gbase + (unsigned long long)kb * 2ull;
        u32x4 g0;
        g0[0] = 1u;                                  // count=1 valid descriptor
        g0[1] = ldsOff;                              // lds_addr
        g0[2] = (unsigned)ga;                        // global_addr[31:0]
        g0[3] = (unsigned)(ga >> 32) | (2u << 30);   // global_addr[56:32] | type=2 (image)
        TDM_LOAD_TO_LDS(g0, g1, gz, gz);
    };

    if (wave == 0) issue_tdm(ldsOff0, 0);            // prologue: slab 0 in flight

    for (int s = 0; s < NSTEPS; ++s) {
        const int kb = s * KSTEP;
        const _Float16* ldsCur = &ldsW[s & 1][0];

        if (wave == 0) __builtin_amdgcn_s_wait_tensorcnt(0);  // slab s landed
        __syncthreads();                                      // publish + retire prev reads
        if (wave == 0 && (s + 1) < NSTEPS)                    // next slab overlaps compute
            issue_tdm((s & 1) ? ldsOff0 : ldsOff1, kb + KSTEP);

        // A fragment: x[m, kb+kOff .. +7] and x[m, kb+kOff+16 .. +23]  (f32 -> f16)
        const float* xp = xrow + kb + kOff;
        f4 a0 = ((const f4*)xp)[0];
        f4 a1 = ((const f4*)xp)[1];
        f4 a2 = ((const f4*)(xp + 16))[0];
        f4 a3 = ((const f4*)(xp + 16))[1];
        __builtin_prefetch(xrow + kb + 2 * KSTEP, 0, 1);      // global_prefetch_b8: x stream

        h4 c0 = __builtin_convertvector(a0, h4);
        h4 c1 = __builtin_convertvector(a1, h4);
        h4 c2 = __builtin_convertvector(a2, h4);
        h4 c3 = __builtin_convertvector(a3, h4);
        h8 lo = __builtin_shufflevector(c0, c1, 0,1,2,3,4,5,6,7);
        h8 hi = __builtin_shufflevector(c2, c3, 0,1,2,3,4,5,6,7);
        v16h afrag = __builtin_shufflevector(lo, hi,
                        0,1,2,3,4,5,6,7,8,9,10,11,12,13,14,15);

#pragma unroll
        for (int nt = 0; nt < 16; ++nt) {
            // B fragment: W1^T[kb+kchunk, n], contiguous k in padded LDS row n
            const _Float16* bp = ldsCur + (nt * 16 + l16) * LDS_STRIDE + kOff;
            h8 b0  = *(const h8*)bp;
            h8 b1v = *(const h8*)(bp + 16);
            v16h bfrag = __builtin_shufflevector(b0, b1v,
                            0,1,2,3,4,5,6,7,8,9,10,11,12,13,14,15);
            acc[nt] = __builtin_amdgcn_wmma_f32_16x16x32_f16(
                          false, afrag, false, bfrag,
                          (short)0, acc[nt], false, false);
        }
    }

    // epilogue: bias + ReLU, store fp16 activations [m, n]
#pragma unroll
    for (int nt = 0; nt < 16; ++nt) {
        const int col = nt * 16 + l16;
        const float bias = sb1[col];
#pragma unroll
        for (int r = 0; r < 8; ++r) {
            const int m = mBase + hlf * 8 + r;   // C/D layout: VGPR r -> M = r + 8*hlf
            float v = acc[nt][r] + bias;
            v = fmaxf(v, 0.0f);
            aout[(size_t)m * H1 + col] = (_Float16)v;
        }
    }
}

// ------------------------------------------------------------------
// Kernel 2: tail MLP  512 -> 32 -> 32 -> 1 (+sigmoid). One thread per position.
// ------------------------------------------------------------------
__global__ __launch_bounds__(256)
void nnue_tail_kernel(const _Float16* __restrict__ a,
                      const float* __restrict__ W2, const float* __restrict__ b2,
                      const float* __restrict__ W3, const float* __restrict__ b3,
                      const float* __restrict__ W4, const float* __restrict__ b4,
                      float* __restrict__ out) {
    __shared__ _Float16 sW2[32 * 512];   // 32 KB
    __shared__ float    sW3[32 * 32];
    __shared__ float    sb2[32], sb3[32], sW4[32];

    const int tid = threadIdx.x;
    for (int i = tid; i < 32 * 512; i += 256) sW2[i] = (_Float16)W2[i];
    for (int i = tid; i < 32 * 32;  i += 256) sW3[i] = W3[i];
    if (tid < 32) { sb2[tid] = b2[tid]; sb3[tid] = b3[tid]; sW4[tid] = W4[tid]; }
    __syncthreads();

    const int p = blockIdx.x * 256 + tid;
    const _Float16* a1 = a + (size_t)p * H1;             // perspective 0
    const _Float16* a2 = a + (size_t)(BATCH + p) * H1;   // perspective 1

    float h2[32];
#pragma unroll 4
    for (int j = 0; j < 32; ++j) {
        float s = sb2[j];
        const _Float16* w = &sW2[j * 512];
        for (int i = 0; i < H1; ++i) s += (float)a1[i] * (float)w[i];
        for (int i = 0; i < H1; ++i) s += (float)a2[i] * (float)w[H1 + i];
        h2[j] = fmaxf(s, 0.0f);
    }

    float o = b4[0];
#pragma unroll
    for (int j = 0; j < 32; ++j) {
        float s = sb3[j];
#pragma unroll
        for (int i = 0; i < 32; ++i) s += sW3[j * 32 + i] * h2[i];
        o += sW4[j] * fmaxf(s, 0.0f);
    }
    out[p] = 1.0f / (1.0f + __expf(-o));
}

// ------------------------------------------------------------------
extern "C" void kernel_launch(void* const* d_in, const int* in_sizes, int n_in,
                              void* d_out, int out_size, void* d_ws, size_t ws_size,
                              hipStream_t stream) {
    const float* x  = (const float*)d_in[0];
    const float* W1 = (const float*)d_in[1];
    const float* b1 = (const float*)d_in[2];
    const float* W2 = (const float*)d_in[3];
    const float* b2 = (const float*)d_in[4];
    const float* W3 = (const float*)d_in[5];
    const float* b3 = (const float*)d_in[6];
    const float* W4 = (const float*)d_in[7];
    const float* b4 = (const float*)d_in[8];
    float* out = (float*)d_out;

    // workspace layout: [W1 fp16: 18.9 MB][activations fp16: 4 MB]
    _Float16* w1h  = (_Float16*)d_ws;
    _Float16* aact = (_Float16*)((char*)d_ws + (size_t)H1 * F_DIM * sizeof(_Float16));

    const int n4 = (H1 * F_DIM) / 4;
    cvt_w1_kernel<<<(n4 + 255) / 256, 256, 0, stream>>>(W1, w1h, n4);
    nnue_layer1_kernel<<<M_TOT / 128, 256, 0, stream>>>(x, w1h, b1, aact);
    nnue_tail_kernel<<<BATCH / 256, 256, 0, stream>>>(aact, W2, b2, W3, b3, W4, b4, out);
}
